// ScaledDotProductAttention_18700287607453
// MI455X (gfx1250) — matrix operand order repaired and checked by reading, not compiled, not run on hardware
//
#include <hip/hip_runtime.h>

typedef __attribute__((ext_vector_type(16))) _Float16 v16h;
typedef __attribute__((ext_vector_type(8)))  _Float16 v8h;
typedef __attribute__((ext_vector_type(2)))  _Float16 h2v;
typedef __attribute__((ext_vector_type(8)))  float    v8f;

constexpr int Bc = 2, Hc = 16, Sc = 2048, DKc = 64, DVc = 64;
constexpr int QTILE = 16;      // q rows per wave
constexpr int KTILE = 32;      // keys per iteration (staged fallback)
constexpr int KT2   = 64;      // keys per iteration (fast path)
constexpr int WAVES_FAST = 8;  // 256 threads, fast path
constexpr int WAVES_STG  = 4;  // 128 threads, staged fallback

union AFrag { v16h v; v8h h[2]; _Float16 e[16]; };
union HPack { h2v h; int i; };

// ---------------------------------------------------------------------------
// Pre-pass: K -> f16 row-major [bh][k][d]; V -> f16 transposed VT[bh][dv][k].
// ~50 MB of traffic once; the 16.8 MB result stays resident in the 192 MB L2
// so the attention kernel's redundant fragment reads never touch HBM.
// ---------------------------------------------------------------------------
__global__ __launch_bounds__(256)
void convert_kv_f16(const float* __restrict__ K, const float* __restrict__ V,
                    _Float16* __restrict__ Kh, _Float16* __restrict__ VT)
{
    const int N = Bc * Hc * Sc * DKc;
    const int i = blockIdx.x * 256 + threadIdx.x;
    if (i >= N) return;
    Kh[i] = (_Float16)K[i];
    const int perhead = Sc * DVc;
    const int bh = i / perhead;
    const int rem = i % perhead;
    const int dv = rem / Sc;
    const int k  = rem % Sc;
    VT[i] = (_Float16)V[(size_t)bh * perhead + (size_t)k * DVc + dv];
}

// ---------------------------------------------------------------------------
// Fast path: flash attention over 64-key tiles. All WMMA B-fragments are
// contiguous 32B/lane global loads from L2-resident f16 buffers. Row-max
// reduction is packed-f16 (stability only -> exact); row-sum is kept as a
// per-lane partial and reduced once at the epilogue.
// ---------------------------------------------------------------------------
__global__ __launch_bounds__(WAVES_FAST * 32)
void sdpa_flash_wmma_direct(const float* __restrict__ Q,
                            const _Float16* __restrict__ Kh,
                            const _Float16* __restrict__ VT,
                            const unsigned char* __restrict__ Msk,
                            float* __restrict__ O)
{
    __shared__ alignas(64) _Float16 Pw[WAVES_FAST][16][KT2];   // 16 KB

    const int tid    = threadIdx.x;
    const int wave   = tid >> 5;
    const int lane   = tid & 31;
    const int laneq  = lane & 15;
    const int laneHi = lane >> 4;
    const int b8     = laneHi ? 8 : 0;

    const int bh = blockIdx.y;
    const int q0 = blockIdx.x * (WAVES_FAST * QTILE) + wave * QTILE;

    const float*         Qh = Q   + (size_t)bh * Sc * DKc;
    const _Float16*      Kp = Kh  + (size_t)bh * Sc * DKc;    // [S][64] f16
    const _Float16*      Vt = VT  + (size_t)bh * DVc * Sc;    // [64][S] f16
    const unsigned char* Mp = Msk + (size_t)bh * Sc * Sc;
    float*               Op = O   + (size_t)bh * Sc * DVc;

    // ---- Q fragments (fp32 -> f16), 16-bit A-matrix layout for 16x16x32 ----
    AFrag aq0, aq1;
    {
        const float* qrow = Qh + (size_t)(q0 + laneq) * DKc;
#pragma unroll
        for (int j = 0; j < 8; ++j) {
            aq0.e[j]     = (_Float16)qrow[2 * b8 + j];
            aq0.e[8 + j] = (_Float16)qrow[2 * b8 + 8 + j];
            aq1.e[j]     = (_Float16)qrow[32 + 2 * b8 + j];
            aq1.e[8 + j] = (_Float16)qrow[32 + 2 * b8 + 8 + j];
        }
    }

    v8f acc[4] = {{}, {}, {}, {}};
    float m_i[8], lp[8];
#pragma unroll
    for (int r = 0; r < 8; ++r) { m_i[r] = -__builtin_inff(); lp[r] = 0.0f; }

    const _Float16* kbase = Kp + (size_t)laneq * DKc + laneHi * 16;
    const _Float16* vb[4] = {
        Vt + (size_t)(laneq     ) * Sc + laneHi * 16,
        Vt + (size_t)(laneq + 16) * Sc + laneHi * 16,
        Vt + (size_t)(laneq + 32) * Sc + laneHi * 16,
        Vt + (size_t)(laneq + 48) * Sc + laneHi * 16 };

    for (int ks = 0; ks < Sc; ks += KT2) {
        // ---- S = Q K^T over 4 key sub-tiles of 16 ----
        const _Float16* kf = kbase + (size_t)ks * DKc;
        v8f s[4];
#pragma unroll
        for (int n = 0; n < 4; ++n) {
            v16h bkA = *(const v16h*)(kf + (size_t)(n * 16) * DKc);
            v16h bkB = *(const v16h*)(kf + (size_t)(n * 16) * DKc + 32);
            v8f z = {};
            z = __builtin_amdgcn_wmma_f32_16x16x32_f16(false, aq0.v, false, bkA, (short)0, z, false, false);
            s[n] = __builtin_amdgcn_wmma_f32_16x16x32_f16(false, aq1.v, false, bkB, (short)0, z, false, false);
        }

        if (ks + KT2 < Sc) {  // gfx1250 global_prefetch of next K tile + mask
            __builtin_prefetch(kf + (size_t)KT2 * DKc, 0, 0);
            __builtin_prefetch(Mp + (size_t)(q0 + laneHi * 8) * Sc + ks + KT2 + laneq, 0, 0);
        }

        // ---- mask (exact, applied to f32 scores) ----
        const unsigned char* mrowbase = Mp + (size_t)(q0 + laneHi * 8) * Sc + ks + laneq;
        float w0[8], w1[8], w2[8], w3[8];
#pragma unroll
        for (int r = 0; r < 8; ++r) {
            const unsigned char* mr = mrowbase + (size_t)r * Sc;
            w0[r] = mr[0]  ? -1e9f : s[0][r];
            w1[r] = mr[16] ? -1e9f : s[1][r];
            w2[r] = mr[32] ? -1e9f : s[2][r];
            w3[r] = mr[48] ? -1e9f : s[3][r];
        }

        // ---- packed-f16 row-max butterfly (2 rows per dword, 16 bpermutes) ----
        HPack d[4];
#pragma unroll
        for (int p = 0; p < 4; ++p) {
            const int ra = 2 * p, rb = 2 * p + 1;
            float ta = fmaxf(fmaxf(w0[ra], w1[ra]), fmaxf(w2[ra], w3[ra]));
            float tb = fmaxf(fmaxf(w0[rb], w1[rb]), fmaxf(w2[rb], w3[rb]));
            ta = fmaxf(ta, -6.0e4f);   // keep finite in f16 (stability bound only)
            tb = fmaxf(tb, -6.0e4f);
            d[p].h = (h2v){ (_Float16)ta, (_Float16)tb };
        }
#pragma unroll
        for (int st = 1; st <= 8; st <<= 1) {
#pragma unroll
            for (int p = 0; p < 4; ++p) {
                HPack o; o.i = __shfl_xor(d[p].i, st);
                d[p].h = __builtin_elementwise_max(d[p].h, o.h);
            }
        }

        // ---- online softmax update; row sums stay lane-local ----
#pragma unroll
        for (int r = 0; r < 8; ++r) {
            const float tmax  = (float)d[r >> 1].h[r & 1];
            const float mnew  = fmaxf(m_i[r], tmax);
            const float alpha = __expf(m_i[r] - mnew);
            m_i[r] = mnew;
            const float p0 = __expf(w0[r] - mnew);
            const float p1 = __expf(w1[r] - mnew);
            const float p2 = __expf(w2[r] - mnew);
            const float p3 = __expf(w3[r] - mnew);
            lp[r] = lp[r] * alpha + ((p0 + p1) + (p2 + p3));
            acc[0][r] *= alpha; acc[1][r] *= alpha;
            acc[2][r] *= alpha; acc[3][r] *= alpha;
            _Float16* prow = &Pw[wave][r + laneHi * 8][laneq];
            prow[0]  = (_Float16)p0;
            prow[16] = (_Float16)p1;
            prow[32] = (_Float16)p2;
            prow[48] = (_Float16)p3;
        }

        // ---- O += P V : two K=32 steps over the 64-key tile ----
        AFrag ap0, ap1;
        ap0.h[0] = *(const v8h*)&Pw[wave][laneq][2 * b8];
        ap0.h[1] = *(const v8h*)&Pw[wave][laneq][2 * b8 + 8];
        ap1.h[0] = *(const v8h*)&Pw[wave][laneq][32 + 2 * b8];
        ap1.h[1] = *(const v8h*)&Pw[wave][laneq][32 + 2 * b8 + 8];
#pragma unroll
        for (int nb = 0; nb < 4; ++nb) {
            v16h bvA = *(const v16h*)(vb[nb] + ks);
            v16h bvB = *(const v16h*)(vb[nb] + ks + 32);
            acc[nb] = __builtin_amdgcn_wmma_f32_16x16x32_f16(false, ap0.v, false, bvA, (short)0, acc[nb], false, false);
            acc[nb] = __builtin_amdgcn_wmma_f32_16x16x32_f16(false, ap1.v, false, bvB, (short)0, acc[nb], false, false);
        }
    }

    // ---- epilogue: one row-sum reduction, normalize, store fp32 ----
#pragma unroll
    for (int r = 0; r < 8; ++r) {
        float l = lp[r];
        l += __shfl_xor(l, 1);
        l += __shfl_xor(l, 2);
        l += __shfl_xor(l, 4);
        l += __shfl_xor(l, 8);
        const float inv = 1.0f / l;
        float* orow = Op + (size_t)(q0 + r + laneHi * 8) * DVc;
#pragma unroll
        for (int nb = 0; nb < 4; ++nb)
            orow[nb * 16 + laneq] = acc[nb][r] * inv;
    }
}

// ---------------------------------------------------------------------------
// Fallback (no workspace): LDS-staged kernel (known-good), unchanged.
// ---------------------------------------------------------------------------
__global__ __launch_bounds__(WAVES_STG * 32)
void sdpa_flash_wmma_staged(const float* __restrict__ Q,
                            const float* __restrict__ K,
                            const float* __restrict__ V,
                            const unsigned char* __restrict__ Msk,
                            float* __restrict__ O)
{
    __shared__ alignas(64) _Float16 Kh[KTILE][DKc];
    __shared__ alignas(64) _Float16 VT[DVc][KTILE];
    __shared__ alignas(64) _Float16 Pw[WAVES_STG][16][KTILE];

    const int tid    = threadIdx.x;
    const int wave   = tid >> 5;
    const int lane   = tid & 31;
    const int laneq  = lane & 15;
    const int laneHi = lane >> 4;
    const int b8     = laneHi ? 8 : 0;

    const int bh = blockIdx.y;
    const int q0 = blockIdx.x * (WAVES_STG * QTILE) + wave * QTILE;

    const float* Qh = Q + (size_t)bh * Sc * DKc;
    const float* Kp = K + (size_t)bh * Sc * DKc;
    const float* Vp = V + (size_t)bh * Sc * DVc;
    const unsigned char* Mp = Msk + (size_t)bh * Sc * Sc;
    float* Op = O + (size_t)bh * Sc * DVc;

    AFrag aq0, aq1;
    {
        const float* qrow = Qh + (size_t)(q0 + laneq) * DKc;
#pragma unroll
        for (int j = 0; j < 8; ++j) {
            aq0.e[j]     = (_Float16)qrow[2 * b8 + j];
            aq0.e[8 + j] = (_Float16)qrow[2 * b8 + 8 + j];
            aq1.e[j]     = (_Float16)qrow[32 + 2 * b8 + j];
            aq1.e[8 + j] = (_Float16)qrow[32 + 2 * b8 + 8 + j];
        }
    }

    v8f acc[4] = {{}, {}, {}, {}};
    float m_i[8], l_i[8];
#pragma unroll
    for (int r = 0; r < 8; ++r) { m_i[r] = -__builtin_inff(); l_i[r] = 0.0f; }

    for (int ks = 0; ks < Sc; ks += KTILE) {
        __syncthreads();
#pragma unroll
        for (int i = 0; i < 4; ++i) {
            const int flat = i * 512 + tid * 4;
            const int row  = flat >> 6;
            const int col  = flat & 63;
            const float4 kv = *(const float4*)(Kp + (size_t)ks * DKc + flat);
            const float4 vv = *(const float4*)(Vp + (size_t)ks * DVc + flat);
            _Float16* kd = &Kh[row][col];
            kd[0] = (_Float16)kv.x; kd[1] = (_Float16)kv.y;
            kd[2] = (_Float16)kv.z; kd[3] = (_Float16)kv.w;
            VT[col + 0][row] = (_Float16)vv.x;
            VT[col + 1][row] = (_Float16)vv.y;
            VT[col + 2][row] = (_Float16)vv.z;
            VT[col + 3][row] = (_Float16)vv.w;
        }
        __syncthreads();

        v16h bk00 = *(const v16h*)&Kh[laneq][     laneHi * 16];
        v16h bk01 = *(const v16h*)&Kh[laneq][32 + laneHi * 16];
        v16h bk10 = *(const v16h*)&Kh[16 + laneq][     laneHi * 16];
        v16h bk11 = *(const v16h*)&Kh[16 + laneq][32 + laneHi * 16];

        v8f s0 = {}, s1 = {};
        s0 = __builtin_amdgcn_wmma_f32_16x16x32_f16(false, aq0.v, false, bk00, (short)0, s0, false, false);
        s0 = __builtin_amdgcn_wmma_f32_16x16x32_f16(false, aq1.v, false, bk01, (short)0, s0, false, false);
        s1 = __builtin_amdgcn_wmma_f32_16x16x32_f16(false, aq0.v, false, bk10, (short)0, s1, false, false);
        s1 = __builtin_amdgcn_wmma_f32_16x16x32_f16(false, aq1.v, false, bk11, (short)0, s1, false, false);

        const unsigned char* mrowbase = Mp + (size_t)(q0 + laneHi * 8) * Sc + ks + laneq;
#pragma unroll
        for (int r = 0; r < 8; ++r) {
            const unsigned char* mr = mrowbase + (size_t)r * Sc;
            float wA = mr[0]  ? -1e9f : s0[r];
            float wB = mr[16] ? -1e9f : s1[r];
            float t = fmaxf(wA, wB);
            t = fmaxf(t, __shfl_xor(t, 1));
            t = fmaxf(t, __shfl_xor(t, 2));
            t = fmaxf(t, __shfl_xor(t, 4));
            t = fmaxf(t, __shfl_xor(t, 8));
            const float mnew  = fmaxf(m_i[r], t);
            const float alpha = __expf(m_i[r] - mnew);
            m_i[r] = mnew;
            const float p0 = __expf(wA - mnew);
            const float p1 = __expf(wB - mnew);
            float ps = p0 + p1;
            ps += __shfl_xor(ps, 1);
            ps += __shfl_xor(ps, 2);
            ps += __shfl_xor(ps, 4);
            ps += __shfl_xor(ps, 8);
            l_i[r] = l_i[r] * alpha + ps;
            acc[0][r] *= alpha; acc[1][r] *= alpha;
            acc[2][r] *= alpha; acc[3][r] *= alpha;
            Pw[wave][r + laneHi * 8][laneq]      = (_Float16)p0;
            Pw[wave][r + laneHi * 8][16 + laneq] = (_Float16)p1;
        }

        AFrag ap;
        ap.h[0] = *(const v8h*)&Pw[wave][laneq][2 * b8];
        ap.h[1] = *(const v8h*)&Pw[wave][laneq][2 * b8 + 8];

#pragma unroll
        for (int nb = 0; nb < 4; ++nb) {
            v16h bv = *(const v16h*)&VT[nb * 16 + laneq][laneHi * 16];
            acc[nb] = __builtin_amdgcn_wmma_f32_16x16x32_f16(false, ap.v, false, bv, (short)0, acc[nb], false, false);
        }
    }

#pragma unroll
    for (int r = 0; r < 8; ++r) {
        const float inv = 1.0f / l_i[r];
        float* orow = Op + (size_t)(q0 + r + laneHi * 8) * DVc;
#pragma unroll
        for (int nb = 0; nb < 4; ++nb)
            orow[nb * 16 + laneq] = acc[nb][r] * inv;
    }
}

extern "C" void kernel_launch(void* const* d_in, const int* in_sizes, int n_in,
                              void* d_out, int out_size, void* d_ws, size_t ws_size,
                              hipStream_t stream) {
    (void)in_sizes; (void)n_in; (void)out_size;
    const float* q = (const float*)d_in[0];
    const float* k = (const float*)d_in[1];
    const float* v = (const float*)d_in[2];
    const unsigned char* m = (const unsigned char*)d_in[3];  // jnp.bool_ = 1 byte
    float* o = (float*)d_out;

    const size_t N    = (size_t)Bc * Hc * Sc * DKc;          // 4.19M elems
    const size_t need = 2 * N * sizeof(_Float16);            // 16.8 MB

    if (d_ws != nullptr && ws_size >= need) {
        _Float16* Kh16 = (_Float16*)d_ws;
        _Float16* VT16 = Kh16 + N;
        hipLaunchKernelGGL(convert_kv_f16, dim3((unsigned)((N + 255) / 256)),
                           dim3(256), 0, stream, k, v, Kh16, VT16);
        dim3 grid(Sc / (WAVES_FAST * QTILE), Bc * Hc);       // (16, 32)
        hipLaunchKernelGGL(sdpa_flash_wmma_direct, grid, dim3(WAVES_FAST * 32),
                           0, stream, q, Kh16, VT16, m, o);
    } else {
        dim3 grid(Sc / (WAVES_STG * QTILE), Bc * Hc);        // (32, 32)
        hipLaunchKernelGGL(sdpa_flash_wmma_staged, grid, dim3(WAVES_STG * 32),
                           0, stream, q, k, v, m, o);
    }
}
